// SDPATransformerLayer_30288109371665
// MI455X (gfx1250) — compile-verified, matrix operand order checked by reading, not run on hardware
//
#include <hip/hip_runtime.h>
#include <hip/hip_bf16.h>
#include <math.h>

// ---------------------------------------------------------------------------
// SDPA transformer layer for MI455X (gfx1250), wave32 + WMMA f16->f32,
// GEMM + attention tiles streamed by the Tensor Data Mover (TDM).
// ---------------------------------------------------------------------------

typedef __attribute__((ext_vector_type(16))) _Float16 v16h;
typedef __attribute__((ext_vector_type(8)))  _Float16 v8h;
typedef __attribute__((ext_vector_type(8)))  float    v8f;

#define BB   2
#define SS   2048
#define DDIM 768
#define HH   12
#define DKK  64
#define DFF  3072
#define MM   (BB * SS)          // 4096 rows total

__device__ __forceinline__ v8f vzero8() {
  v8f z = {0.f, 0.f, 0.f, 0.f, 0.f, 0.f, 0.f, 0.f};
  return z;
}

// Load a 16-lane-striped 16x32 f16 WMMA operand fragment from a row-major
// tile in LDS. Per ISA 7.12.2 (16-bit A/B operand layout, wave32):
//   lanes 0-15 : K = [0,8) in v0..v3, K = [16,24) in v4..v7
//   lanes 16-31: K = [8,16)          K = [24,32)
__device__ __forceinline__ v16h frag_ld(const _Float16* __restrict__ base,
                                        int row, int ld, int lane) {
  const _Float16* p = base + (size_t)row * ld + ((lane & 16) ? 8 : 0);
  v8h lo = *(const v8h*)(p);
  v8h hi = *(const v8h*)(p + 16);
  v16h r;
#pragma unroll
  for (int i = 0; i < 8; ++i) { r[i] = lo[i]; r[8 + i] = hi[i]; }
  return r;
}

__device__ __forceinline__ v8f wmma_f16(v16h a, v16h b, v8f c) {
  return __builtin_amdgcn_wmma_f32_16x16x32_f16(
      /*neg_a=*/false, a, /*neg_b=*/false, b,
      /*c_mod=*/(short)0, c, /*reuse_a=*/false, /*reuse_b=*/false);
}

// ---------------------------------------------------------------------------
// Tensor Data Mover: DMA a (tileH x tileW)-half tile (row stride =
// strideElems halves) from global into LDS, appending 4 DWORDs (8 halves) of
// padding after every 2<<padIntLog DWORDs (= one tile row), producing an LDS
// row stride of tileW + 8 halves. Descriptor per CDNA5 ISA.
// ---------------------------------------------------------------------------
#if __has_builtin(__builtin_amdgcn_tensor_load_to_lds)
#define ATHENA_TDM 1
typedef unsigned int v4u __attribute__((ext_vector_type(4)));
typedef int          v8i __attribute__((ext_vector_type(8)));
typedef int          v4i __attribute__((ext_vector_type(4)));

__device__ __forceinline__ void tdm_wait0() {
#if __has_builtin(__builtin_amdgcn_s_wait_tensorcnt)
  __builtin_amdgcn_s_wait_tensorcnt(0);
#else
  asm volatile("s_wait_tensorcnt 0x0" ::: "memory");
#endif
}

__device__ __forceinline__ void tdm_tile_load(
    const _Float16* gsrc, const _Float16* ldst, unsigned strideElems,
    unsigned tileW, unsigned tileH, unsigned padIntLog) {
  const unsigned long long ga = (unsigned long long)(uintptr_t)gsrc;
  const unsigned           la = (unsigned)(uintptr_t)ldst;  // LDS byte address
  // group0: count=1 | lds_addr | global_addr[56:0] | type=2 in bits 127:126
  v4u g0 = { 1u, la, (unsigned)ga,
             (unsigned)((ga >> 32) & 0x01FFFFFFull) | 0x80000000u };
  v8i g1;
  g1[0] = (int)((1u << 16)              // data_size = 1 (2-byte elements)
        | (1u << 20)                    // pad_enable
        | (padIntLog << 22)             // pad after 2<<padIntLog DWORDs
        | (3u << 25));                  // pad_amount: 4 DWORDs (8 halves)
  g1[1] = (int)((strideElems & 0xFFFFu) << 16);  // tensor_dim0[15:0]
  g1[2] = (int)0xFFFF0000u;   // tensor_dim0 hi = 0, tensor_dim1 lo = 65535
  g1[3] = (int)(tileW << 16); // tensor_dim1 hi = 0, tile_dim0
  g1[4] = (int)tileH;         // tile_dim1, tile_dim2 = 0 (2D)
  g1[5] = (int)strideElems;   // tensor_dim0_stride[31:0]
  g1[6] = 0;
  g1[7] = 0;
  v4i gz = {0, 0, 0, 0};
#if defined(__clang_major__) && (__clang_major__ >= 23)
  v8i gz8 = {0, 0, 0, 0, 0, 0, 0, 0};
  __builtin_amdgcn_tensor_load_to_lds(g0, g1, gz, gz, gz8, 0);
#else
  __builtin_amdgcn_tensor_load_to_lds(g0, g1, gz, gz, 0);
#endif
}
#endif  // ATHENA_TDM

// ---------------------------------------------------------------------------
// Weight convert: fp32 row-major (K x N) -> f16 transposed (N x K).
// ---------------------------------------------------------------------------
__global__ __launch_bounds__(256) void convertT_kernel(
    const float* __restrict__ in, _Float16* __restrict__ out, int K, int N) {
  size_t i = (size_t)blockIdx.x * 256 + threadIdx.x;
  if (i >= (size_t)K * N) return;
  int k = (int)(i / N), n = (int)(i % N);
  out[(size_t)n * K + k] = (_Float16)in[i];
}

// ---------------------------------------------------------------------------
// LayerNorm: one 256-thread block per row of 768; f32 + f16 outputs.
// ---------------------------------------------------------------------------
__global__ __launch_bounds__(256) void ln_kernel(
    const float* __restrict__ x, const float* __restrict__ g,
    const float* __restrict__ b, float* __restrict__ y32,
    _Float16* __restrict__ y16) {
  const int row = blockIdx.x;
  const float* xr = x + (size_t)row * DDIM;
  float s = 0.f, s2 = 0.f;
  for (int i = threadIdx.x; i < DDIM; i += 256) {
    float v = xr[i];
    s += v; s2 += v * v;
  }
#pragma unroll
  for (int off = 16; off >= 1; off >>= 1) {
    s  += __shfl_xor(s,  off, 32);
    s2 += __shfl_xor(s2, off, 32);
  }
  __shared__ float red[16];
  const int wave = threadIdx.x >> 5, lane = threadIdx.x & 31;
  if (lane == 0) { red[wave] = s; red[8 + wave] = s2; }
  __syncthreads();
  float ts = 0.f, t2 = 0.f;
#pragma unroll
  for (int w = 0; w < 8; ++w) { ts += red[w]; t2 += red[8 + w]; }
  const float mean = ts * (1.0f / DDIM);
  const float var  = t2 * (1.0f / DDIM) - mean * mean;
  const float rstd = rsqrtf(var + 1e-5f);
  for (int i = threadIdx.x; i < DDIM; i += 256) {
    float v = (xr[i] - mean) * rstd * g[i] + b[i];
    size_t idx = (size_t)row * DDIM + i;
    y32[idx] = v;
    y16[idx] = (_Float16)v;
  }
}

// ---------------------------------------------------------------------------
// WMMA GEMM: C(M x N) = A(M x K, f16 rm) * BT(N x K, f16 rm)^T.
// Block tile 128x128, K-step 32, 256 threads = 8 waves in a 4x2 grid:
// wave (wr,wc) owns a 32x64 sub-tile = 2 A-frags x 4 B-frags -> 8 WMMA.
// K-tiles are DMA'd by the TDM into double-buffered padded LDS tiles;
// tile k+1 is in flight while tile k is consumed by the matrix pipes.
// EPI: 0 = store f16, 1 = f32 out = acc + resid(f32), 2 = f16 out = gelu(acc)
// ---------------------------------------------------------------------------
#define GLDT 40   // LDS tile stride in halves (32 + 8 pad; 80B rows)

template <int EPI>
__global__ __launch_bounds__(256) void gemm_kernel(
    const _Float16* __restrict__ A, const _Float16* __restrict__ BT,
    int Mdim, int Ndim, int Kdim, _Float16* __restrict__ o16,
    float* __restrict__ o32, const float* __restrict__ resid) {
  (void)Mdim;
  __shared__ _Float16 sA[2][128 * GLDT];
  __shared__ _Float16 sB[2][128 * GLDT];
  const int tid = threadIdx.x, wave = tid >> 5, lane = tid & 31;
  const int wr = wave >> 1, wc = wave & 1;          // 4x2 wave grid
  const int bm = blockIdx.y * 128, bn = blockIdx.x * 128;

  v8f acc[2][4];
#pragma unroll
  for (int mi = 0; mi < 2; ++mi)
#pragma unroll
    for (int ni = 0; ni < 4; ++ni) acc[mi][ni] = vzero8();

  const int ktiles = Kdim >> 5;

#ifdef ATHENA_TDM
  if (wave == 0) {   // prologue: DMA first K-tile into buffer 0
    tdm_tile_load(A  + (size_t)bm * Kdim, sA[0], (unsigned)Kdim, 32, 128, 3);
    tdm_tile_load(BT + (size_t)bn * Kdim, sB[0], (unsigned)Kdim, 32, 128, 3);
  }
#endif

  for (int kt = 0; kt < ktiles; ++kt) {
#ifdef ATHENA_TDM
    const int buf = kt & 1;
    if (wave == 0) tdm_wait0();       // current buffer landed in LDS
    __syncthreads();                  // publish; also WAR-guards alt buffer
    if (wave == 0 && (kt + 1) < ktiles) {
      const size_t k0n = (size_t)(kt + 1) << 5;
      tdm_tile_load(A  + (size_t)bm * Kdim + k0n, sA[buf ^ 1],
                    (unsigned)Kdim, 32, 128, 3);
      tdm_tile_load(BT + (size_t)bn * Kdim + k0n, sB[buf ^ 1],
                    (unsigned)Kdim, 32, 128, 3);
    }
#else
    const int buf = 0;
    const int k0 = kt << 5;
    __syncthreads();
#pragma unroll
    for (int c = tid; c < 512; c += 256) {
      const int r = c >> 2, kc = (c & 3) * 8;
      const _Float16* ap = A  + ((size_t)(bm + r) * Kdim + k0 + kc);
      const _Float16* bp = BT + ((size_t)(bn + r) * Kdim + k0 + kc);
      *(v8h*)(sA[0] + r * GLDT + kc) = *(const v8h*)ap;
      *(v8h*)(sB[0] + r * GLDT + kc) = *(const v8h*)bp;
      __builtin_prefetch(ap + 32, 0, 1);
      __builtin_prefetch(bp + 32, 0, 1);
    }
    __syncthreads();
#endif

    v16h af[2], bf[4];
#pragma unroll
    for (int mi = 0; mi < 2; ++mi)
      af[mi] = frag_ld(sA[buf], wr * 32 + mi * 16 + (lane & 15), GLDT, lane);
#pragma unroll
    for (int ni = 0; ni < 4; ++ni)
      bf[ni] = frag_ld(sB[buf], wc * 64 + ni * 16 + (lane & 15), GLDT, lane);
#pragma unroll
    for (int mi = 0; mi < 2; ++mi)
#pragma unroll
      for (int ni = 0; ni < 4; ++ni)
        acc[mi][ni] = wmma_f16(af[mi], bf[ni], acc[mi][ni]);
  }

  // C/D layout: n = lane&15 (+16*ni), m = vgpr + 8*(lane>=16).
  const int hi8 = (lane & 16) ? 8 : 0;
  const int n0  = bn + wc * 64 + (lane & 15);
#pragma unroll
  for (int mi = 0; mi < 2; ++mi) {
    const int mbase = bm + wr * 32 + mi * 16 + hi8;
#pragma unroll
    for (int ni = 0; ni < 4; ++ni) {
#pragma unroll
      for (int vr = 0; vr < 8; ++vr) {
        const size_t idx = (size_t)(mbase + vr) * Ndim + n0 + ni * 16;
        const float v = acc[mi][ni][vr];
        if (EPI == 0) {
          o16[idx] = (_Float16)v;
        } else if (EPI == 1) {
          o32[idx] = v + resid[idx];
        } else {
          const float gelu = 0.5f * v * (1.0f + erff(v * 0.70710678118654752f));
          o16[idx] = (_Float16)gelu;
        }
      }
    }
  }
}

// ---------------------------------------------------------------------------
// Flash attention (causal + count_bias + softcap). Grid: (S/64, B*H).
// 128 threads = 4 waves; wave w owns q rows [16w,16w+16) of the 64-row block.
// Q and K tiles are DMA'd by the TDM (row-major, padded to ALDT); the K DMA
// is issued before the manual V-transpose staging so it overlaps that work.
// ---------------------------------------------------------------------------
#define ALDT 72   // 64 + 8 pad halves; 144B rows

__global__ __launch_bounds__(128) void attn_kernel(
    const _Float16* __restrict__ Q, const _Float16* __restrict__ Km,
    const _Float16* __restrict__ V, const float* __restrict__ bias,
    _Float16* __restrict__ O) {
  const int qt = blockIdx.x;
  const int bh = blockIdx.y;
  const int b = bh / HH, h = bh % HH;
  const int tid = threadIdx.x, wave = tid >> 5, lane = tid & 31;

  __shared__ _Float16 sQ [64 * ALDT];
  __shared__ _Float16 sK [64 * ALDT];     // [kv][dk]  (B^T for Q@K^T)
  __shared__ _Float16 sVT[64 * ALDT];     // [dk][kv]  (B^T for P@V)
  __shared__ _Float16 sP [4 * 16 * ALDT]; // per-wave probability tiles
  _Float16* sPw = sP + wave * 16 * ALDT;

  const size_t qrow0 = (size_t)b * SS + (size_t)qt * 64;
#ifdef ATHENA_TDM
  if (wave == 0) {
    tdm_tile_load(Q + qrow0 * DDIM + h * DKK, sQ, DDIM, 64, 64, 4);
    tdm_wait0();
  }
#else
#pragma unroll
  for (int c = tid; c < 512; c += 128) {
    const int r = c >> 3, kc = (c & 7) * 8;
    *(v8h*)(sQ + r * ALDT + kc) =
        *(const v8h*)(Q + (qrow0 + r) * DDIM + h * DKK + kc);
  }
#endif
  __syncthreads();

  v16h qf[2];
  qf[0] = frag_ld(sQ + 0,  wave * 16 + (lane & 15), ALDT, lane);
  qf[1] = frag_ld(sQ + 32, wave * 16 + (lane & 15), ALDT, lane);

  v8f oacc[4];
#pragma unroll
  for (int i = 0; i < 4; ++i) oacc[i] = vzero8();
  float mrun[8], lrun[8];
#pragma unroll
  for (int i = 0; i < 8; ++i) { mrun[i] = -1e30f; lrun[i] = 0.f; }

  const float scale = 0.125f;             // 1/sqrt(64)
  const int mloc = wave * 16 + ((lane & 16) ? 8 : 0);

  for (int kt = 0; kt <= qt; ++kt) {
    __syncthreads();                      // waves done reading sK/sVT
    const size_t krow0 = (size_t)b * SS + (size_t)kt * 64;
#ifdef ATHENA_TDM
    if (wave == 0)                        // K DMA overlaps V staging below
      tdm_tile_load(Km + krow0 * DDIM + h * DKK, sK, DDIM, 64, 64, 4);
#pragma unroll
    for (int c = tid; c < 512; c += 128) {
      const int r = c >> 3, kc = (c & 7) * 8;
      const v8h vv = *(const v8h*)(V + (krow0 + r) * DDIM + h * DKK + kc);
#pragma unroll
      for (int j = 0; j < 8; ++j) sVT[(kc + j) * ALDT + r] = vv[j];
    }
    if (wave == 0) tdm_wait0();
#else
#pragma unroll
    for (int c = tid; c < 512; c += 128) {
      const int r = c >> 3, kc = (c & 7) * 8;
      *(v8h*)(sK + r * ALDT + kc) =
          *(const v8h*)(Km + (krow0 + r) * DDIM + h * DKK + kc);
      const v8h vv = *(const v8h*)(V + (krow0 + r) * DDIM + h * DKK + kc);
#pragma unroll
      for (int j = 0; j < 8; ++j) sVT[(kc + j) * ALDT + r] = vv[j];
    }
#endif
    __syncthreads();

    // scores: S = Q @ K^T (16x64 per wave)
    v8f sacc[4];
#pragma unroll
    for (int i = 0; i < 4; ++i) sacc[i] = vzero8();
#pragma unroll
    for (int ct = 0; ct < 4; ++ct) {
#pragma unroll
      for (int ks = 0; ks < 2; ++ks) {
        const v16h kf = frag_ld(sK + ks * 32, ct * 16 + (lane & 15), ALDT, lane);
        sacc[ct] = wmma_f16(qf[ks], kf, sacc[ct]);
      }
    }

    // scale + count_bias + softcap + causal mask
    const int mglob0 = qt * 64 + mloc;
    const int nbase  = kt * 64 + (lane & 15);
#pragma unroll
    for (int ct = 0; ct < 4; ++ct) {
      const int n = nbase + ct * 16;
#pragma unroll
      for (int vr = 0; vr < 8; ++vr) {
        const int m = mglob0 + vr;
        float s = sacc[ct][vr] * scale + bias[((size_t)b * SS + m) * SS + n];
        s = tanhf(s * (1.0f / 30.0f)) * 30.0f;
        sacc[ct][vr] = (n <= m) ? s : -1e30f;
      }
    }

    // online softmax (rows live across the 16-lane half-groups)
#pragma unroll
    for (int vr = 0; vr < 8; ++vr) {
      float rowm = -1e30f;
#pragma unroll
      for (int ct = 0; ct < 4; ++ct) rowm = fmaxf(rowm, sacc[ct][vr]);
#pragma unroll
      for (int off = 1; off < 16; off <<= 1)
        rowm = fmaxf(rowm, __shfl_xor(rowm, off, 16));
      const float mnew = fmaxf(mrun[vr], rowm);
      const float sf = expf(mrun[vr] - mnew);
      mrun[vr] = mnew;
      float rs = 0.f;
#pragma unroll
      for (int ct = 0; ct < 4; ++ct) {
        const float p = expf(sacc[ct][vr] - mnew);
        sacc[ct][vr] = p;
        rs += p;
      }
#pragma unroll
      for (int off = 1; off < 16; off <<= 1) rs += __shfl_xor(rs, off, 16);
      lrun[vr] = lrun[vr] * sf + rs;
#pragma unroll
      for (int ct = 0; ct < 4; ++ct) oacc[ct][vr] *= sf;
    }

    // restage P (C-layout regs -> row-major LDS, per-wave region)
    {
      const int r0 = (lane & 16) ? 8 : 0;
      const int c0 = lane & 15;
#pragma unroll
      for (int ct = 0; ct < 4; ++ct)
#pragma unroll
        for (int vr = 0; vr < 8; ++vr)
          sPw[(r0 + vr) * ALDT + ct * 16 + c0] = (_Float16)sacc[ct][vr];
    }

    // O += P @ V
    v16h pf[2];
    pf[0] = frag_ld(sPw + 0,  lane & 15, ALDT, lane);
    pf[1] = frag_ld(sPw + 32, lane & 15, ALDT, lane);
#pragma unroll
    for (int ctd = 0; ctd < 4; ++ctd) {
#pragma unroll
      for (int ks = 0; ks < 2; ++ks) {
        const v16h vf = frag_ld(sVT + ks * 32, ctd * 16 + (lane & 15), ALDT, lane);
        oacc[ctd] = wmma_f16(pf[ks], vf, oacc[ctd]);
      }
    }
  }

  // normalize and store O (f16, (B,S,H,DK) row-major == M x D)
#pragma unroll
  for (int ctd = 0; ctd < 4; ++ctd) {
#pragma unroll
    for (int vr = 0; vr < 8; ++vr) {
      const int m = qt * 64 + mloc + vr;
      const size_t idx =
          ((size_t)b * SS + m) * DDIM + h * DKK + ctd * 16 + (lane & 15);
      O[idx] = (_Float16)(oacc[ctd][vr] / lrun[vr]);
    }
  }
}

// ---------------------------------------------------------------------------
// Host launcher
// ---------------------------------------------------------------------------
extern "C" void kernel_launch(void* const* d_in, const int* in_sizes, int n_in,
                              void* d_out, int out_size, void* d_ws,
                              size_t ws_size, hipStream_t stream) {
  (void)in_sizes; (void)n_in; (void)out_size; (void)ws_size;
  const float* x  = (const float*)d_in[0];
  const float* cb = (const float*)d_in[1];
  // d_in[2]: attn_mask — causal, applied analytically in-kernel.
  const float* Wq = (const float*)d_in[3];
  const float* Wk = (const float*)d_in[4];
  const float* Wv = (const float*)d_in[5];
  const float* Wo = (const float*)d_in[6];
  const float* W1 = (const float*)d_in[7];
  const float* W2 = (const float*)d_in[8];
  const float* g1 = (const float*)d_in[9];
  const float* b1 = (const float*)d_in[10];
  const float* g2 = (const float*)d_in[11];
  const float* b2 = (const float*)d_in[12];
  float* out = (float*)d_out;

  char* w = (char*)d_ws;
  auto alloc = [&](size_t bytes) -> char* {
    char* p = w;
    w += (bytes + 255) & ~(size_t)255;
    return p;
  };
  float*    h32   = (float*)   alloc((size_t)MM * DDIM * 4);
  _Float16* h16   = (_Float16*)alloc((size_t)MM * DDIM * 2);
  _Float16* WqT   = (_Float16*)alloc((size_t)DDIM * DDIM * 2);
  _Float16* WkT   = (_Float16*)alloc((size_t)DDIM * DDIM * 2);
  _Float16* WvT   = (_Float16*)alloc((size_t)DDIM * DDIM * 2);
  _Float16* WoT   = (_Float16*)alloc((size_t)DDIM * DDIM * 2);
  _Float16* W1T   = (_Float16*)alloc((size_t)DDIM * DFF * 2);
  _Float16* W2T   = (_Float16*)alloc((size_t)DFF * DDIM * 2);
  _Float16* q16   = (_Float16*)alloc((size_t)MM * DDIM * 2);
  _Float16* k16   = (_Float16*)alloc((size_t)MM * DDIM * 2);
  _Float16* v16   = (_Float16*)alloc((size_t)MM * DDIM * 2);
  _Float16* o16   = (_Float16*)alloc((size_t)MM * DDIM * 2);
  float*    x1    = (float*)   alloc((size_t)MM * DDIM * 4);
  float*    h2_32 = (float*)   alloc((size_t)MM * DDIM * 4);
  _Float16* h2_16 = (_Float16*)alloc((size_t)MM * DDIM * 2);
  _Float16* f1    = (_Float16*)alloc((size_t)MM * DFF * 2);

  // 1) weights -> f16 transposed
  {
    const int nsq = (DDIM * DDIM + 255) / 256;
    convertT_kernel<<<nsq, 256, 0, stream>>>(Wq, WqT, DDIM, DDIM);
    convertT_kernel<<<nsq, 256, 0, stream>>>(Wk, WkT, DDIM, DDIM);
    convertT_kernel<<<nsq, 256, 0, stream>>>(Wv, WvT, DDIM, DDIM);
    convertT_kernel<<<nsq, 256, 0, stream>>>(Wo, WoT, DDIM, DDIM);
    const int nff = (DDIM * DFF + 255) / 256;
    convertT_kernel<<<nff, 256, 0, stream>>>(W1, W1T, DDIM, DFF);
    convertT_kernel<<<nff, 256, 0, stream>>>(W2, W2T, DFF, DDIM);
  }

  // 2) LN1
  ln_kernel<<<MM, 256, 0, stream>>>(x, g1, b1, h32, h16);

  // 3) QKV projections
  const dim3 gD(DDIM / 128, MM / 128);   // (6, 32)
  gemm_kernel<0><<<gD, 256, 0, stream>>>(h16, WqT, MM, DDIM, DDIM, q16, nullptr, nullptr);
  gemm_kernel<0><<<gD, 256, 0, stream>>>(h16, WkT, MM, DDIM, DDIM, k16, nullptr, nullptr);
  gemm_kernel<0><<<gD, 256, 0, stream>>>(h16, WvT, MM, DDIM, DDIM, v16, nullptr, nullptr);

  // 4) attention
  attn_kernel<<<dim3(SS / 64, BB * HH), 128, 0, stream>>>(q16, k16, v16, cb, o16);

  // 5) output projection + residual (x1 = o@Wo + h)
  gemm_kernel<1><<<gD, 256, 0, stream>>>(o16, WoT, MM, DDIM, DDIM, nullptr, x1, h32);

  // 6) LN2
  ln_kernel<<<MM, 256, 0, stream>>>(x1, g2, b2, h2_32, h2_16);

  // 7) FFN up + exact GELU
  gemm_kernel<2><<<dim3(DFF / 128, MM / 128), 256, 0, stream>>>(
      h2_16, W1T, MM, DFF, DDIM, f1, nullptr, nullptr);

  // 8) FFN down + residual from h2 (reference: return ffn + h2)
  gemm_kernel<1><<<gD, 256, 0, stream>>>(f1, W2T, MM, DDIM, DFF, nullptr, out, h2_32);
}
